// HNEMA_metapath_specific_78245714199240
// MI455X (gfx1250) — matrix operand order repaired and checked by reading, not compiled, not run on hardware
//
#include <hip/hip_runtime.h>

typedef __attribute__((ext_vector_type(16))) _Float16 v16h;
typedef __attribute__((ext_vector_type(8)))  _Float16 v8h;
typedef __attribute__((ext_vector_type(8)))  float    v8f;

#define HPAD 264   // 256 + 8 halves pad -> row stride 528B, avoids LDS bank conflicts

// ---------------------------------------------------------------- WMMA helpers
__device__ __forceinline__ v8f wmma_f16(v16h a, v16h b, v8f c) {
  // D = A(16x32 f16) * B(32x16 f16) + C(16x16 f32)
  return __builtin_amdgcn_wmma_f32_16x16x32_f16(false, a, false, b, (short)0, c,
                                                false, false);
}

// A fragment (16x32 f16) from a row-major row of >=k0+32 halves.
// lane holds row (lane&15); hi = lane>>4.
// elems 0..7  -> K = k0 + hi*8 + e
// elems 8..15 -> K = k0 + 16 + hi*8 + (e-8)
__device__ __forceinline__ v16h load_afrag(const _Float16* row, int k0, int hi) {
  v8h lo = *(const v8h*)(row + k0 + hi * 8);
  v8h hh = *(const v8h*)(row + k0 + 16 + hi * 8);
  return __builtin_shufflevector(lo, hh, 0,1,2,3,4,5,6,7,8,9,10,11,12,13,14,15);
}

// B fragment from prepacked weights: [frag][lane][16 halves] contiguous.
__device__ __forceinline__ v16h load_bfrag(const _Float16* __restrict__ wp,
                                           int frag, int lane) {
  return *(const v16h*)(wp + ((size_t)frag << 9) + (lane << 4));
}

__device__ __forceinline__ float sigmoidf_(float x) {
  return 1.0f / (1.0f + __expf(-x));
}

// ---------------------------------------------------------------- prep kernels
// Pack W (R x K, row-major f32) into WMMA-B fragment layout (f16):
// frag f = nt*(K/32)+kt ; element (lane,e): W[nt*16 + (lane&15), kt*32 + 16*(e>>3) + 8*(lane>>4) + (e&7)]
__global__ void pack_wT_kernel(const float* __restrict__ W, _Float16* __restrict__ dst,
                               int R, int K) {
  int t = blockIdx.x * 256 + threadIdx.x;
  if (t >= R * K) return;
  int e = t & 15;
  int l = (t >> 4) & 31;
  int f = t >> 9;
  int numKt = K >> 5;
  int kt = f % numKt;
  int nt = f / numKt;
  int row = nt * 16 + (l & 15);
  int col = kt * 32 + ((e >> 3) << 4) + ((l >> 4) << 3) + (e & 7);
  dst[t] = (_Float16)W[(size_t)row * K + col];
}

__global__ void f32_to_f16_kernel(const float* __restrict__ src,
                                  _Float16* __restrict__ dst, int n) {
  int t = blockIdx.x * 256 + threadIdx.x;
  if (t < n) dst[t] = (_Float16)src[t];
}

__global__ void zero_f32_kernel(float* __restrict__ p, long long n) {
  long long t = (long long)blockIdx.x * 256 + threadIdx.x;
  if (t < n) p[t] = 0.0f;
}

__global__ void init_nodes_kernel(float* __restrict__ m, float* __restrict__ den, int n) {
  int t = blockIdx.x * 256 + threadIdx.x;
  if (t < n) { m[t] = -3.0e38f; den[t] = 0.0f; }
}

// ---------------------------------------------------------------- main GRU + attn
// 1 wave per block, 32 edges per wave (2 M-tiles): every B fragment feeds 2 WMMAs,
// halving weight-operand bandwidth per FLOP vs a 16-edge tile.
__global__ __launch_bounds__(32)
void hnema_gru_attn_kernel(const _Float16* __restrict__ feat16,
                           const _Float16* __restrict__ wih_f, const _Float16* __restrict__ whh_f,
                           const float*   __restrict__ bih_f, const float*   __restrict__ bhh_f,
                           const _Float16* __restrict__ wih_b, const _Float16* __restrict__ whh_b,
                           const float*   __restrict__ bih_b, const float*   __restrict__ bhh_b,
                           const float*   __restrict__ attn,
                           const int*     __restrict__ emi,     // (E,4)
                           _Float16* __restrict__ eft16,        // (E,8,64) f16
                           float*    __restrict__ a_ws,         // (E,8) logits
                           int E) {
  __shared__ _Float16 Hs[2][32][HPAD];   // ping-pong hidden state, 32 edge rows
  __shared__ float    As[32][8];         // attention logits accum

  const int lane = threadIdx.x;
  const int rowm = lane & 15;
  const int hi   = lane >> 4;
  const int e0   = blockIdx.x * 32;
  const int eA   = e0 + rowm;            // tile 0 edge
  const int eB   = e0 + 16 + rowm;       // tile 1 edge
  const int eeA  = eA < E ? eA : (E - 1);
  const int eeB  = eB < E ? eB : (E - 1);

  for (int p = lane; p < 256; p += 32) As[p >> 3][p & 7] = 0.0f;

#pragma unroll 1
  for (int dir = 0; dir < 2; ++dir) {
    const _Float16* wih = dir ? wih_b : wih_f;
    const _Float16* whh = dir ? whh_b : whh_f;
    const float*    bih = dir ? bih_b : bih_f;
    const float*    bhh = dir ? bhh_b : bhh_f;

    // h0 = 0
    {
      _Float16* hz = &Hs[0][0][0];
      for (int i = lane; i < 32 * HPAD; i += 32) hz[i] = (_Float16)0.0f;
    }
    int cur = 0;

#pragma unroll 1
    for (int tt = 0; tt < 4; ++tt) {
      const int t = dir ? (3 - tt) : tt;

      // gathered x fragments for this timestep (per-lane row pointers), both tiles
      const _Float16* xrow0 = feat16 + (size_t)emi[(size_t)eeA * 4 + t] * 64;
      const _Float16* xrow1 = feat16 + (size_t)emi[(size_t)eeB * 4 + t] * 64;
      const v16h xa00 = load_afrag(xrow0, 0, hi);   // tile0, k0=0
      const v16h xa01 = load_afrag(xrow0, 32, hi);  // tile0, k0=32
      const v16h xa10 = load_afrag(xrow1, 0, hi);   // tile1, k0=0
      const v16h xa11 = load_afrag(xrow1, 32, hi);  // tile1, k0=32

#pragma unroll 1
      for (int c = 0; c < 16; ++c) {
        v8f ar0 = {}, az0 = {}, ain0 = {}, ahn0 = {};  // tile 0
        v8f ar1 = {}, az1 = {}, ain1 = {}, ahn1 = {};  // tile 1

        // gi parts (K=64, numKt=2): gate row-tiles r:c, z:16+c, n:32+c
        {
          v16h br0 = load_bfrag(wih, (c)      * 2 + 0, lane);
          v16h bz0 = load_bfrag(wih, (16 + c) * 2 + 0, lane);
          v16h bn0 = load_bfrag(wih, (32 + c) * 2 + 0, lane);
          ar0  = wmma_f16(xa00, br0, ar0);  ar1  = wmma_f16(xa10, br0, ar1);
          az0  = wmma_f16(xa00, bz0, az0);  az1  = wmma_f16(xa10, bz0, az1);
          ain0 = wmma_f16(xa00, bn0, ain0); ain1 = wmma_f16(xa10, bn0, ain1);
          v16h br1 = load_bfrag(wih, (c)      * 2 + 1, lane);
          v16h bz1 = load_bfrag(wih, (16 + c) * 2 + 1, lane);
          v16h bn1 = load_bfrag(wih, (32 + c) * 2 + 1, lane);
          ar0  = wmma_f16(xa01, br1, ar0);  ar1  = wmma_f16(xa11, br1, ar1);
          az0  = wmma_f16(xa01, bz1, az0);  az1  = wmma_f16(xa11, bz1, az1);
          ain0 = wmma_f16(xa01, bn1, ain0); ain1 = wmma_f16(xa11, bn1, ain1);
        }
        // gh parts (K=256, numKt=8); h fragments re-read from LDS per k-step
#pragma unroll
        for (int kk = 0; kk < 8; ++kk) {
          v16h h0 = load_afrag(&Hs[cur][rowm][0],      kk * 32, hi);
          v16h h1 = load_afrag(&Hs[cur][16 + rowm][0], kk * 32, hi);
          v16h br = load_bfrag(whh, (c)      * 8 + kk, lane);
          v16h bz = load_bfrag(whh, (16 + c) * 8 + kk, lane);
          v16h bn = load_bfrag(whh, (32 + c) * 8 + kk, lane);
          ar0  = wmma_f16(h0, br, ar0);   ar1  = wmma_f16(h1, br, ar1);
          az0  = wmma_f16(h0, bz, az0);   az1  = wmma_f16(h1, bz, az1);
          ahn0 = wmma_f16(h0, bn, ahn0);  ahn1 = wmma_f16(h1, bn, ahn1);
        }

        // elementwise GRU cell update for this 16-column chunk (both tiles)
        const int n   = c * 16 + (lane & 15);       // hidden index, this lane's column
        const float br_  = bih[n] + bhh[n];
        const float bz_  = bih[256 + n] + bhh[256 + n];
        const float bin_ = bih[512 + n];
        const float bhn_ = bhh[512 + n];
#pragma unroll
        for (int r8 = 0; r8 < 8; ++r8) {
          const int m = r8 + hi * 8;               // edge row within tile (C/D layout)
          // tile 0
          {
            const float rg = sigmoidf_(ar0[r8] + br_);
            const float zg = sigmoidf_(az0[r8] + bz_);
            const float ng = tanhf(ain0[r8] + bin_ + rg * (ahn0[r8] + bhn_));
            const float ho = (float)Hs[cur][m][n];
            Hs[cur ^ 1][m][n] = (_Float16)((1.0f - zg) * ng + zg * ho);
          }
          // tile 1
          {
            const float rg = sigmoidf_(ar1[r8] + br_);
            const float zg = sigmoidf_(az1[r8] + bz_);
            const float ng = tanhf(ain1[r8] + bin_ + rg * (ahn1[r8] + bhn_));
            const float ho = (float)Hs[cur][16 + m][n];
            Hs[cur ^ 1][16 + m][n] = (_Float16)((1.0f - zg) * ng + zg * ho);
          }
        }
      }
      cur ^= 1;
    }

    // epilogue: Hs[cur] = final hidden of this direction (32 x 256)
    // eft[e,h,d] = hidden[e, d*8 + h]; forward covers d in [0,32), backward [32,64)
    for (int i = lane; i < 32 * 256; i += 32) {
      const int row = i >> 8;
      const int j   = i & 255;
      const int d   = (j >> 3) + dir * 32;
      const int h   = j & 7;
      if (e0 + row < E)
        eft16[(size_t)(e0 + row) * 512 + h * 64 + d] = Hs[cur][row][j];
    }
    // partial attention logits: a[row,h] += sum_d h[row, d*8+h] * attn[h,d]
    for (int p = lane; p < 256; p += 32) {
      const int row = p >> 3;
      const int h   = p & 7;
      float s = 0.0f;
      for (int dd = 0; dd < 32; ++dd)
        s += (float)Hs[cur][row][dd * 8 + h] * attn[h * 64 + dd + dir * 32];
      As[row][h] += s;
    }
  }

  // LeakyReLU and store logits
  for (int p = lane; p < 256; p += 32) {
    const int row = p >> 3;
    const int h   = p & 7;
    if (e0 + row < E) {
      const float aa = As[row][h];
      a_ws[(size_t)(e0 + row) * 8 + h] = aa >= 0.0f ? aa : 0.01f * aa;
    }
  }
}

// ---------------------------------------------------------------- segment softmax
__device__ __forceinline__ void atomicMaxF(float* addr, float val) {
  int* ai = (int*)addr;
  int cur = __float_as_int(*addr);
  while (__int_as_float(cur) < val) {
    int prev = atomicCAS(ai, cur, __float_as_int(val));
    if (prev == cur) break;
    cur = prev;
  }
}

__global__ void seg_max_kernel(const float* __restrict__ a_ws, const int* __restrict__ dst,
                               float* __restrict__ m, int E) {
  int t = blockIdx.x * 256 + threadIdx.x;
  if (t >= E * 8) return;
  int e = t >> 3, h = t & 7;
  atomicMaxF(&m[(size_t)dst[e] * 8 + h], a_ws[t]);
}

__global__ void seg_expsum_kernel(float* __restrict__ a_ws, const int* __restrict__ dst,
                                  const float* __restrict__ m, float* __restrict__ den, int E) {
  int t = blockIdx.x * 256 + threadIdx.x;
  if (t >= E * 8) return;
  int e = t >> 3, h = t & 7;
  float ex = __expf(a_ws[t] - m[(size_t)dst[e] * 8 + h]);
  a_ws[t] = ex;                                 // in-place: logits -> exp
  atomicAdd(&den[(size_t)dst[e] * 8 + h], ex);
}

__global__ void scatter_kernel(const _Float16* __restrict__ eft16,
                               const float* __restrict__ ex_ws,
                               const float* __restrict__ den,
                               const int* __restrict__ dst,
                               float* __restrict__ out, int E) {
  long long t = (long long)blockIdx.x * 256 + threadIdx.x;
  if (t >= (long long)E * 512) return;
  int e = (int)(t >> 9);
  int j = (int)(t & 511);
  int h = j >> 6;
  int n = dst[e];
  float alpha = ex_ws[(size_t)e * 8 + h] / den[(size_t)n * 8 + h];
  atomicAdd(out + (size_t)n * 512 + j, alpha * (float)eft16[t]);
}

// ---------------------------------------------------------------- launcher
extern "C" void kernel_launch(void* const* d_in, const int* in_sizes, int n_in,
                              void* d_out, int out_size, void* d_ws, size_t ws_size,
                              hipStream_t stream) {
  const float* features = (const float*)d_in[0];
  const float* w_ih_f   = (const float*)d_in[1];
  const float* w_hh_f   = (const float*)d_in[2];
  const float* b_ih_f   = (const float*)d_in[3];
  const float* b_hh_f   = (const float*)d_in[4];
  const float* w_ih_b   = (const float*)d_in[5];
  const float* w_hh_b   = (const float*)d_in[6];
  const float* b_ih_b   = (const float*)d_in[7];
  const float* b_hh_b   = (const float*)d_in[8];
  const float* attn     = (const float*)d_in[9];
  const int*   emi      = (const int*)d_in[10];
  const int*   edst     = (const int*)d_in[11];
  float*       out      = (float*)d_out;

  const int E = in_sizes[11];          // 500000
  const int N = in_sizes[0] / 64;      // 100000

  // workspace carving
  const size_t hw1 = (size_t)768 * 64;   // w_ih pack (halves)
  const size_t hw2 = (size_t)768 * 256;  // w_hh pack (halves)
  _Float16* wp_ih_f = (_Float16*)d_ws;
  _Float16* wp_hh_f = wp_ih_f + hw1;
  _Float16* wp_ih_b = wp_hh_f + hw2;
  _Float16* wp_hh_b = wp_ih_b + hw1;
  _Float16* feat16  = wp_hh_b + hw2;
  _Float16* eft16   = feat16 + (size_t)N * 64;
  float*    a_ws    = (float*)(eft16 + (size_t)E * 512);
  float*    m_ws    = a_ws + (size_t)E * 8;
  float*    den_ws  = m_ws + (size_t)N * 8;

  // 1) pack weights to WMMA-B fragment layout (f16)
  pack_wT_kernel<<<(768 * 64  + 255) / 256, 256, 0, stream>>>(w_ih_f, wp_ih_f, 768, 64);
  pack_wT_kernel<<<(768 * 256 + 255) / 256, 256, 0, stream>>>(w_hh_f, wp_hh_f, 768, 256);
  pack_wT_kernel<<<(768 * 64  + 255) / 256, 256, 0, stream>>>(w_ih_b, wp_ih_b, 768, 64);
  pack_wT_kernel<<<(768 * 256 + 255) / 256, 256, 0, stream>>>(w_hh_b, wp_hh_b, 768, 256);

  // 2) features -> f16
  f32_to_f16_kernel<<<(N * 64 + 255) / 256, 256, 0, stream>>>(features, feat16, N * 64);

  // 3) init output and segment-softmax state
  const long long outElems = (long long)N * 512;
  zero_f32_kernel<<<(unsigned)((outElems + 255) / 256), 256, 0, stream>>>(out, outElems);
  init_nodes_kernel<<<(N * 8 + 255) / 256, 256, 0, stream>>>(m_ws, den_ws, N * 8);

  // 4) bi-GRU + attention logits (WMMA core): 1 wave / 32 edges per block
  hnema_gru_attn_kernel<<<(E + 31) / 32, 32, 0, stream>>>(
      feat16, wp_ih_f, wp_hh_f, b_ih_f, b_hh_f,
      wp_ih_b, wp_hh_b, b_ih_b, b_hh_b,
      attn, emi, eft16, a_ws, E);

  // 5) segment softmax
  seg_max_kernel<<<(E * 8 + 255) / 256, 256, 0, stream>>>(a_ws, edst, m_ws, E);
  seg_expsum_kernel<<<(E * 8 + 255) / 256, 256, 0, stream>>>(a_ws, edst, m_ws, den_ws, E);

  // 6) weighted scatter into output
  const long long sc = (long long)E * 512;
  scatter_kernel<<<(unsigned)((sc + 255) / 256), 256, 0, stream>>>(
      eft16, a_ws, den_ws, edst, out, E);
}
